// LNN_35107062678332
// MI455X (gfx1250) — matrix operand order repaired and checked by reading, not compiled, not run on hardware
//
#include <hip/hip_runtime.h>

// ---------------- types / constants ----------------
typedef __attribute__((ext_vector_type(16))) __bf16 v16bf;
typedef __attribute__((ext_vector_type(8)))  float  v8f;
typedef __attribute__((ext_vector_type(4)))  float  f32x4;

#define K_ACT  0.5f
#define A_COEF 1.0f

// workspace layout (ushort/bf16 elements)
#define WS_W1B_M   0        // 64x256
#define WS_W1TB_M  16384    // 256x64
#define WS_W2B_M   32768    // 256x256
#define WS_W2TB_M  98304    // 256x256
#define WS_W1B_Q   163840
#define WS_W1TB_Q  180224
#define WS_W2B_Q   196608
#define WS_W2TB_Q  262144
#define WS_TOTAL_US 327680

__device__ __forceinline__ unsigned short f2bf(float f) {
  union { float f; unsigned u; } c; c.f = f;
  unsigned u = c.u;
  unsigned r = u + 0x7FFFu + ((u >> 16) & 1u);   // round-to-nearest-even
  return (unsigned short)(r >> 16);
}
__device__ __forceinline__ float bf2f(unsigned short h) {
  union { unsigned u; float f; } c; c.u = ((unsigned)h) << 16;
  return c.f;
}

// ---- WMMA fragment loaders (CDNA5 ISA 7.12.2, wave32, 16-bit operands) ----
// A operand: 16x32 (MxK), source row-major with contiguous K (stride ld).
// lanes 0-15 hold K 0-7 & 16-23 of row M=lane; lanes 16-31 hold K 8-15 & 24-31.
__device__ __forceinline__ v16bf load_fragA(const unsigned short* base, int ld,
                                            int m0, int k0, int lane) {
  int m = m0 + (lane & 15);
  int half = lane >> 4;
  const unsigned short* p = base + m * ld + k0 + 8 * half;
  v16bf f;
  f32x4* fv = reinterpret_cast<f32x4*>(&f);
  fv[0] = *reinterpret_cast<const f32x4*>(p);        // VGPR0-3: 8 contiguous K
  fv[1] = *reinterpret_cast<const f32x4*>(p + 16);   // VGPR4-7: next K group
  return f;
}
// B operand: 32x16 (KxN), source stored as Q^T row-major (row n holds contiguous K).
// lanes 0-15 hold K 0-7 & 16-23 of column N=lane; lanes 16-31 hold K 8-15 & 24-31.
__device__ __forceinline__ v16bf load_fragB(const unsigned short* base, int ld,
                                            int n0, int k0, int lane) {
  int n = n0 + (lane & 15);
  int half = lane >> 4;
  const unsigned short* p = base + n * ld + k0 + 8 * half;
  v16bf f;
  f32x4* fv = reinterpret_cast<f32x4*>(&f);
  fv[0] = *reinterpret_cast<const f32x4*>(p);
  fv[1] = *reinterpret_cast<const f32x4*>(p + 16);
  return f;
}

// ---------------- prep: fp32 weights -> bf16 (+ transposes) ----------------
__global__ void lnn_prep_kernel(const float* __restrict__ mW1, const float* __restrict__ mW2,
                                const float* __restrict__ qW1, const float* __restrict__ qW2,
                                unsigned short* __restrict__ ws) {
  int idx = blockIdx.x * blockDim.x + threadIdx.x;   // 0..65535
  if (idx < 16384) {                                 // W1: 64x256
    int i = idx >> 8, j = idx & 255;
    unsigned short vm = f2bf(mW1[idx]);
    ws[WS_W1B_M  + idx]          = vm;
    ws[WS_W1TB_M + j * 64 + i]   = vm;
    unsigned short vq = f2bf(qW1[idx]);
    ws[WS_W1B_Q  + idx]          = vq;
    ws[WS_W1TB_Q + j * 64 + i]   = vq;
  }
  if (idx < 65536) {                                 // W2: 256x256
    int k = idx >> 8, n = idx & 255;
    unsigned short vm = f2bf(mW2[idx]);
    ws[WS_W2B_M  + idx]          = vm;
    ws[WS_W2TB_M + n * 256 + k]  = vm;
    unsigned short vq = f2bf(qW2[idx]);
    ws[WS_W2B_Q  + idx]          = vq;
    ws[WS_W2TB_Q + n * 256 + k]  = vq;
  }
}

// ---------------- main: one workgroup (8 wave32) per batch element ----------------
__global__ __launch_bounds__(256)
void lnn_dyn_kernel(const float* __restrict__ x,
                    const float* __restrict__ mb1, const float* __restrict__ mb2,
                    const float* __restrict__ mw3,
                    const float* __restrict__ qb1, const float* __restrict__ qb2,
                    const float* __restrict__ qw3,
                    const unsigned short* __restrict__ wsb,
                    float* __restrict__ out) {
  extern __shared__ __align__(16) char smem[];
  unsigned short* W1s = (unsigned short*)smem;   // 64x256 bf16: W1*diag(s'(z1))
  unsigned short* Cb  = W1s + 64 * 256;          // 64x256 bf16: C = W1s @ W2
  unsigned short* Pb  = Cb  + 64 * 256;          // 32x256 bf16: scaled operand
  float* acth = (float*)(Pb + 32 * 256);         // 256: layer-1 activation
  float* sA   = acth + 256;                      // 256: s'(z1)
  float* sApp = sA   + 256;                      // 256: s''(z1)
  float* uu   = sApp + 256;                      // 256: s'(z2)*w3
  float* w2u  = uu   + 256;                      // 256: W2 @ u
  float* D1   = w2u  + 256;                      // 256: s''(z1)*(W2u)
  float* D2   = D1   + 256;                      // 256: s''(z2)*w3
  float* Hrows= D2   + 256;                      // 32x64: H[32:64, 0:64]
  float* Mrow = Hrows+ 32 * 64;                  // 32x33 augmented solve matrix
  float* sol  = Mrow + 32 * 33;                  // 32
  float* xs   = sol  + 32;                       // 64
  float* grad = xs   + 64;                       // 64
  int*   piv  = (int*)(grad + 64);

  const int tid  = threadIdx.x;
  const int lane = tid & 31;
  const int wave = tid >> 5;
  const int b    = blockIdx.x;

  const unsigned short* W1b[2]  = { wsb + WS_W1B_M,  wsb + WS_W1B_Q  };
  const unsigned short* W1Tb[2] = { wsb + WS_W1TB_M, wsb + WS_W1TB_Q };
  const unsigned short* W2b[2]  = { wsb + WS_W2B_M,  wsb + WS_W2B_Q  };
  const unsigned short* W2Tb[2] = { wsb + WS_W2TB_M, wsb + WS_W2TB_Q };
  const float* b1p[2] = { mb1, qb1 };
  const float* b2p[2] = { mb2, qb2 };
  const float* w3p[2] = { mw3, qw3 };

  if (tid < 64) {
    float xv = x[b * 64 + tid];
    xs[tid]   = xv;
    grad[tid] = (tid >= 32) ? 2.0f * A_COEF * xv : 0.0f;   // quad-term gradient
  }
  __syncthreads();

  // Each wave owns one 16x16 tile of H[32:64, 0:64]; accumulate across GEMM2/3, both paths.
  const int tmH = wave >> 2, tnH = wave & 3;
  v8f Hacc = {0.f,0.f,0.f,0.f,0.f,0.f,0.f,0.f};

  for (int path = 0; path < 2; ++path) {
    const unsigned short* w1  = W1b[path];
    const unsigned short* w1t = W1Tb[path];
    const unsigned short* w2  = W2b[path];
    const unsigned short* w2t = W2Tb[path];
    const float* bias1 = b1p[path];
    const float* bias2 = b2p[path];
    const float* w3    = w3p[path];

    // ---- phase 1: z1[j] = x . W1[:,j] + b1[j]; act + derivatives ----
    {
      int j = tid;
      float acc = bias1[j];
      #pragma unroll 8
      for (int i = 0; i < 64; ++i) acc += xs[i] * bf2f(w1[i * 256 + j]);
      if (path == 0) {                        // softplus
        float s = 1.0f / (1.0f + expf(-acc));
        acth[j] = (acc > 20.0f) ? acc : log1pf(expf(acc));
        sA[j]   = s;                          // s'
        sApp[j] = s * (1.0f - s);             // s''
      } else {                                // a(z) = z*tanh(k z)
        float t = tanhf(K_ACT * acc);
        float sech2 = 1.0f - t * t;
        acth[j] = acc * t;
        sA[j]   = t + K_ACT * acc * sech2;
        sApp[j] = 2.0f * K_ACT * sech2 * (1.0f - K_ACT * acc * t);
      }
    }
    __syncthreads();
    // ---- phase 2: z2[j] = h . W2[:,j] + b2[j]; u, D2 ----
    {
      int j = tid;
      float acc = bias2[j];
      #pragma unroll 8
      for (int k = 0; k < 256; ++k) acc += acth[k] * bf2f(w2[k * 256 + j]);
      float w3j = w3[j];
      if (path == 0) {
        float s = 1.0f / (1.0f + expf(-acc));
        uu[j] = s * w3j;
        D2[j] = s * (1.0f - s) * w3j;
      } else {
        float t = tanhf(K_ACT * acc);
        float sech2 = 1.0f - t * t;
        uu[j] = (t + K_ACT * acc * sech2) * w3j;
        D2[j] = 2.0f * K_ACT * sech2 * (1.0f - K_ACT * acc * t) * w3j;
      }
    }
    __syncthreads();
    // ---- phase 3: (W2 @ u)[j] via W2^T column j; D1 ----
    {
      int j = tid;
      float acc = 0.0f;
      #pragma unroll 8
      for (int k = 0; k < 256; ++k) acc += uu[k] * bf2f(w2t[k * 256 + j]);
      w2u[j] = acc;
      D1[j]  = sApp[j] * acc;
    }
    __syncthreads();
    // ---- phase 4: grad += W1 @ (s'(z1) * W2u) (via W1^T, threads 0..63) ----
    if (tid < 64) {
      float acc = 0.0f;
      #pragma unroll 8
      for (int k = 0; k < 256; ++k) acc += bf2f(w1t[k * 64 + tid]) * (sA[k] * w2u[k]);
      grad[tid] += acc;
    }
    // ---- phase 5: W1s[i,j] = W1[i,j] * s'(z1_j)  (bf16, LDS) ----
    for (int idx = tid; idx < 64 * 256; idx += 256) {
      int j = idx & 255;
      W1s[idx] = f2bf(bf2f(w1[idx]) * sA[j]);
    }
    __syncthreads();
    // ---- GEMM1: C(64x256) = W1s @ W2   (B operand = W2^T rows, global, L2-resident) ----
    for (int t = wave; t < 64; t += 8) {
      int tm = t & 3, tn = t >> 2;
      v8f acc = {0.f,0.f,0.f,0.f,0.f,0.f,0.f,0.f};
      #pragma unroll
      for (int kb = 0; kb < 8; ++kb) {
        v16bf a  = load_fragA(W1s, 256, tm * 16, kb * 32, lane);
        v16bf bb = load_fragB(w2t, 256, tn * 16, kb * 32, lane);
        acc = __builtin_amdgcn_wmma_f32_16x16x32_bf16(false, a, false, bb,
                                                      (short)0, acc, false, false);
      }
      int n = lane & 15, half = lane >> 4;
      #pragma unroll
      for (int v = 0; v < 8; ++v)
        Cb[(tm * 16 + v + 8 * half) * 256 + tn * 16 + n] = f2bf(acc[v]);
    }
    __syncthreads();
    // ---- P[r,j] = C[32+r, j] * D2[j] ----
    for (int idx = tid; idx < 32 * 256; idx += 256) {
      int j = idx & 255;
      Pb[idx] = f2bf(bf2f(Cb[32 * 256 + idx]) * D2[j]);
    }
    __syncthreads();
    // ---- GEMM2: H[32:,:] += (C[32:,:]*D2) @ C^T   (B operand = C rows, LDS) ----
    #pragma unroll
    for (int kb = 0; kb < 8; ++kb) {
      v16bf a  = load_fragA(Pb, 256, tmH * 16, kb * 32, lane);
      v16bf bb = load_fragB(Cb, 256, tnH * 16, kb * 32, lane);
      Hacc = __builtin_amdgcn_wmma_f32_16x16x32_bf16(false, a, false, bb,
                                                     (short)0, Hacc, false, false);
    }
    __syncthreads();
    // ---- A3[r,j] = W1[32+r, j] * D1[j]  (reuse Pb) ----
    for (int idx = tid; idx < 32 * 256; idx += 256) {
      int j = idx & 255;
      Pb[idx] = f2bf(bf2f(w1[32 * 256 + idx]) * D1[j]);
    }
    __syncthreads();
    // ---- GEMM3: H[32:,:] += (W1[32:,:]*D1) @ W1^T  (B operand = W1 rows, global) ----
    #pragma unroll
    for (int kb = 0; kb < 8; ++kb) {
      v16bf a  = load_fragA(Pb, 256, tmH * 16, kb * 32, lane);
      v16bf bb = load_fragB(w1, 256, tnH * 16, kb * 32, lane);
      Hacc = __builtin_amdgcn_wmma_f32_16x16x32_bf16(false, a, false, bb,
                                                     (short)0, Hacc, false, false);
    }
    __syncthreads();
  }

  // ---- write Hessian rows (+ 2*A_COEF on the qq diagonal) ----
  {
    int n = lane & 15, half = lane >> 4;
    #pragma unroll
    for (int v = 0; v < 8; ++v) {
      int r = tmH * 16 + v + 8 * half;     // H row (32+r)
      int c = tnH * 16 + n;                // H col
      float add = (c == 32 + r) ? 2.0f * A_COEF : 0.0f;
      Hrows[r * 64 + c] = Hacc[v] + add;
    }
  }
  __syncthreads();

  // ---- rhs = jac[:32] - B @ qdot ; augmented matrix [A | rhs] ----
  if (tid < 32) {
    float r = grad[tid];
    for (int c = 0; c < 32; ++c) r -= Hrows[tid * 64 + c] * xs[32 + c];
    for (int c = 0; c < 32; ++c) Mrow[tid * 33 + c] = Hrows[tid * 64 + 32 + c];
    Mrow[tid * 33 + 32] = r;
  }
  __syncthreads();

  // ---- Gaussian elimination with partial pivoting (A has +2I from quad term) ----
  for (int k = 0; k < 32; ++k) {
    if (tid == 0) {
      int pi = k; float best = fabsf(Mrow[k * 33 + k]);
      for (int i = k + 1; i < 32; ++i) {
        float v = fabsf(Mrow[i * 33 + k]);
        if (v > best) { best = v; pi = i; }
      }
      *piv = pi;
    }
    __syncthreads();
    int pi = *piv;
    if (pi != k && tid <= 32) {
      float tsw = Mrow[k * 33 + tid];
      Mrow[k * 33 + tid]  = Mrow[pi * 33 + tid];
      Mrow[pi * 33 + tid] = tsw;
    }
    __syncthreads();
    if (tid < 32 && tid > k) {
      float f = Mrow[tid * 33 + k] / Mrow[k * 33 + k];
      for (int j = k; j <= 32; ++j) Mrow[tid * 33 + j] -= f * Mrow[k * 33 + j];
    }
    __syncthreads();
  }
  if (tid == 0) {
    for (int k = 31; k >= 0; --k) {
      float s = Mrow[k * 33 + 32];
      for (int j = k + 1; j < 32; ++j) s -= Mrow[k * 33 + j] * sol[j];
      sol[k] = s / Mrow[k * 33 + k];
    }
  }
  __syncthreads();

  if (tid < 64) out[b * 64 + tid] = (tid < 32) ? xs[32 + tid] : sol[tid - 32];
}

// ---------------- host launcher ----------------
static constexpr size_t LNN_SMEM_BYTES =
    (16384 + 16384 + 8192) * sizeof(unsigned short)             // bf16 staging
    + (7 * 256 + 32 * 64 + 32 * 33 + 32 + 64 + 64) * sizeof(float)
    + 16;                                                       // piv + pad

extern "C" void kernel_launch(void* const* d_in, const int* in_sizes, int n_in,
                              void* d_out, int out_size, void* d_ws, size_t ws_size,
                              hipStream_t stream) {
  const float* x   = (const float*)d_in[0];
  const float* mW1 = (const float*)d_in[1];
  const float* mb1 = (const float*)d_in[2];
  const float* mW2 = (const float*)d_in[3];
  const float* mb2 = (const float*)d_in[4];
  const float* mW3 = (const float*)d_in[5];
  const float* qW1 = (const float*)d_in[6];
  const float* qb1 = (const float*)d_in[7];
  const float* qW2 = (const float*)d_in[8];
  const float* qb2 = (const float*)d_in[9];
  const float* qW3 = (const float*)d_in[10];
  unsigned short* ws = (unsigned short*)d_ws;
  float* out = (float*)d_out;

  int B = in_sizes[0] / 64;
  if (B <= 0) return;

  lnn_prep_kernel<<<256, 256, 0, stream>>>(mW1, mW2, qW1, qW2, ws);
  lnn_dyn_kernel<<<B, 256, LNN_SMEM_BYTES, stream>>>(x, mb1, mb2, mW3,
                                                     qb1, qb2, qW3, ws, out);
}
